// NewtonNetInteraction_59622736003302
// MI455X (gfx1250) — matrix operand order repaired and checked by reading, not compiled, not run on hardware
//
#include <hip/hip_runtime.h>

#define H 128
#define RDIM 32

typedef __attribute__((ext_vector_type(16))) __bf16 v16bf;
typedef __attribute__((ext_vector_type(8)))  float  v8f;

union FragU { v16bf v; uint4 u[2]; };

__device__ __forceinline__ unsigned short f2bf(float x) {
  // round-to-nearest-even f32 -> bf16
  unsigned u = __float_as_uint(x);
  u += 0x7FFFu + ((u >> 16) & 1u);
  return (unsigned short)(u >> 16);
}

__device__ __forceinline__ float silu(float x) { return x / (1.0f + __expf(-x)); }

__device__ __forceinline__ void lds_fence() {
  asm volatile("s_wait_dscnt 0" ::: "memory");
}
__device__ __forceinline__ void async_fence() {
  asm volatile("s_wait_asynccnt 0" ::: "memory");
}

// Async L2->LDS copy of 16B per lane (GLOBAL_LOAD_ASYNC_TO_LDS_B128).
// lds_addr: 32-bit LDS byte address (low 32 bits of flat shared pointer).
__device__ __forceinline__ void async_copy_b128(unsigned lds_addr, const void* gptr) {
  unsigned long long ga = (unsigned long long)(size_t)gptr;
  asm volatile("global_load_async_to_lds_b128 %0, %1, off"
               :: "v"(lds_addr), "v"(ga) : "memory");
}

// A fragment, 16-bit 16x32 (ISA 7.12.2): lane L holds row M=L&15.
// VGPR v, half j -> K = (v>=4?16:0) + ((L>>4)&1)*8 + (v&3)*2 + j
// => two contiguous 8-half (16B) groups at K = hi*8 and K = 16 + hi*8.
__device__ __forceinline__ v16bf load_a_frag(const unsigned short* tile, int strideH, int kc, int lane) {
  const int m  = lane & 15;
  const int hi = (lane >> 4) & 1;
  const uint4* p = (const uint4*)(tile + (size_t)m * strideH + kc * 32 + hi * 8);
  FragU f;
  f.u[0] = p[0];   // K rel = hi*8 .. hi*8+7
  f.u[1] = p[2];   // K rel = 16+hi*8 .. 16+hi*8+7
  return f.v;
}

// B fragment, 16-bit 32x16: lane L holds column N=L&15; K groups hi*8..hi*8+7
// and 16+hi*8..16+hi*8+7 (same as A). Weights stored transposed [N][K] bf16.
__device__ __forceinline__ v16bf load_b_frag(const unsigned short* __restrict__ Wt, int Ktot, int kc, int nt, int lane) {
  const int n  = (lane & 15) + nt * 16;
  const int kg = ((lane >> 4) & 1) * 8;
  const uint4* p = (const uint4*)(Wt + (size_t)n * Ktot + kc * 32 + kg);
  FragU f;
  f.u[0] = p[0];
  f.u[1] = p[2];
  return f.v;
}

__device__ __forceinline__ v8f wmma_bf16(v16bf a, v16bf b, v8f c) {
  return __builtin_amdgcn_wmma_f32_16x16x32_bf16(false, a, false, b, (short)0, c, false, false);
}

__device__ __forceinline__ void zeroC(v8f C[2][8]) {
  const v8f z = {0.f, 0.f, 0.f, 0.f, 0.f, 0.f, 0.f, 0.f};
#pragma unroll
  for (int mt = 0; mt < 2; ++mt)
#pragma unroll
    for (int nt = 0; nt < 8; ++nt) C[mt][nt] = z;
}

// (32 x KTOT) @ (KTOT x 128): two 16-row A tiles share each B fragment, so
// every B load feeds two independent back-to-back WMMAs (hides B latency).
template <int KTOT>
__device__ __forceinline__ void gemm2tile(const unsigned short* aTile, int strideH,
                                          const unsigned short* __restrict__ Wt,
                                          int lane, v8f C[2][8]) {
  for (int kc = 0; kc < KTOT / 32; ++kc) {
    v16bf a0 = load_a_frag(aTile, strideH, kc, lane);
    v16bf a1 = load_a_frag(aTile + 16 * strideH, strideH, kc, lane);
#pragma unroll
    for (int nt = 0; nt < 8; ++nt) {
      v16bf b = load_b_frag(Wt, KTOT, kc, nt, lane);
      C[0][nt] = wmma_bf16(a0, b, C[0][nt]);
      C[1][nt] = wmma_bf16(a1, b, C[1][nt]);
    }
  }
}

// bias + SiLU on f32 accumulators, write 32x128 bf16 h-tile to LDS.
__device__ __forceinline__ void silu_store(unsigned short* hTile, const float* __restrict__ bias,
                                           v8f C[2][8], int lane) {
  const int lnN = lane & 15, hi8 = ((lane >> 4) & 1) * 8;
#pragma unroll
  for (int nt = 0; nt < 8; ++nt) {
    int ncol = nt * 16 + lnN;
    float bb = bias[ncol];
#pragma unroll
    for (int mt = 0; mt < 2; ++mt)
#pragma unroll
      for (int r = 0; r < 8; ++r)
        hTile[(size_t)(mt * 16 + r + hi8) * 128 + ncol] = f2bf(silu(C[mt][nt][r] + bb));
  }
}

// ---------------- prep kernels ----------------

__global__ void wt_convert(const float* __restrict__ W, unsigned short* __restrict__ Wt, int K, int N) {
  int i = blockIdx.x * 256 + threadIdx.x;
  if (i >= K * N) return;
  int k = i / N, n = i - k * N;
  Wt[(size_t)n * K + k] = f2bf(W[i]);
}

__global__ void a2bf(const float* __restrict__ A, unsigned short* __restrict__ B, int n) {
  int i = blockIdx.x * 256 + threadIdx.x;
  if (i < n) B[i] = f2bf(A[i]);
}

// ---------------- edge kernel ----------------
// 2 waves/block, 32 edges/wave. msg and vec MLPs run as sequential passes
// (A tiles re-read from LDS) so accumulators stay at 128 VGPRs.
// Scatter with f32 global atomics (accumulators are L2-resident).

__global__ void __launch_bounds__(64) edge_kernel(
    const float* __restrict__ force_node, const float* __restrict__ dir_edge,
    const float* __restrict__ dist_edge,  const long long* __restrict__ edge_index,
    const unsigned short* __restrict__ atom_bf,
    const unsigned short* __restrict__ WmT1, const unsigned short* __restrict__ WvT1,
    const unsigned short* __restrict__ WmT2, const unsigned short* __restrict__ WvT2,
    const float* __restrict__ b_msg1, const float* __restrict__ b_msg2,
    const float* __restrict__ b_vec1, const float* __restrict__ b_vec2,
    float* __restrict__ scalar_agg, float* __restrict__ vec_agg, int E) {
  __shared__ __align__(16) unsigned short sA[2][32][160];  // src_feat (32 edges x 160)
  __shared__ __align__(16) unsigned short sH[2][32][128];  // hidden tile
  __shared__ int   sRow[2][32];
  __shared__ int   sCol[2][32];
  __shared__ float sDir[2][32][3];

  const int lane = threadIdx.x & 31;
  const int w    = threadIdx.x >> 5;
  const int e0   = (blockIdx.x * 2 + w) * 32;
  if (e0 + 32 > E) return;

  {
    const int e = e0 + lane;
    sRow[w][lane] = (int)edge_index[e];
    sCol[w][lane] = (int)edge_index[(size_t)E + e];
    sDir[w][lane][0] = dir_edge[(size_t)e * 3 + 0];
    sDir[w][lane][1] = dir_edge[(size_t)e * 3 + 1];
    sDir[w][lane][2] = dir_edge[(size_t)e * 3 + 2];
  }
  lds_fence();

  // async gather: 32 rows x 256B of bf16 atom features, straight to LDS
  {
    const unsigned ldsbase = (unsigned)(size_t)&sA[w][0][0];
    for (int t = lane; t < 512; t += 32) {
      int m = t >> 4, q = t & 15;
      const char* g = (const char*)atom_bf + ((size_t)sCol[w][m] * 128) * 2 + q * 16;
      async_copy_b128(ldsbase + m * 320 + q * 16, g);
    }
  }
  // dist features f32 -> bf16 (cols 128..159), overlaps with async copies
  for (int t = lane; t < 1024; t += 32) {
    int m = t >> 5, k = t & 31;
    sA[w][m][128 + k] = f2bf(dist_edge[(size_t)(e0 + m) * RDIM + k]);
  }
  async_fence();
  lds_fence();

  const int lnN = lane & 15, hi8 = ((lane >> 4) & 1) * 8;
  v8f C[2][8];

  // ---- msg path ----
  zeroC(C);
  gemm2tile<160>(&sA[w][0][0], 160, WmT1, lane, C);
  silu_store(&sH[w][0][0], b_msg1, C, lane);
  lds_fence();
  zeroC(C);
  gemm2tile<128>(&sH[w][0][0], 128, WmT2, lane, C);
  {
    float bm2[8];
#pragma unroll
    for (int nt = 0; nt < 8; ++nt) bm2[nt] = b_msg2[nt * 16 + lnN];
#pragma unroll
    for (int mt = 0; mt < 2; ++mt)
#pragma unroll
      for (int r = 0; r < 8; ++r) {
        int M = mt * 16 + r + hi8;
        float* dst = scalar_agg + (size_t)sRow[w][M] * H + lnN;
#pragma unroll
        for (int nt = 0; nt < 8; ++nt)
          atomicAdd(dst + nt * 16, C[mt][nt][r] + bm2[nt]);
      }
  }

  // ---- vec path ----
  zeroC(C);
  gemm2tile<160>(&sA[w][0][0], 160, WvT1, lane, C);
  silu_store(&sH[w][0][0], b_vec1, C, lane);  // msg layer2 done; safe to overwrite
  lds_fence();
  zeroC(C);
  gemm2tile<128>(&sH[w][0][0], 128, WvT2, lane, C);
  {
    float bv2[8];
#pragma unroll
    for (int nt = 0; nt < 8; ++nt) bv2[nt] = b_vec2[nt * 16 + lnN];
#pragma unroll
    for (int mt = 0; mt < 2; ++mt)
#pragma unroll
      for (int r = 0; r < 8; ++r) {
        int M = mt * 16 + r + hi8;
        const int row = sRow[w][M], col = sCol[w][M];
        const float d0 = sDir[w][M][0], d1 = sDir[w][M][1], d2 = sDir[w][M][2];
        const float* fsrc = force_node + (size_t)col * 3 * H + lnN;
        float* vdst = vec_agg + (size_t)row * 3 * H + lnN;
#pragma unroll
        for (int nt = 0; nt < 8; ++nt) {
          int o = nt * 16;
          float vw = C[mt][nt][r] + bv2[nt];
          atomicAdd(vdst + o,         vw * (d0 + fsrc[o]));
          atomicAdd(vdst + H + o,     vw * (d1 + fsrc[H + o]));
          atomicAdd(vdst + 2 * H + o, vw * (d2 + fsrc[2 * H + o]));
        }
      }
  }
}

// ---------------- node kernel ----------------
__global__ void __launch_bounds__(64) node_kernel(
    const float* __restrict__ atom_node, const float* __restrict__ force_node,
    const unsigned short* __restrict__ atom_bf, const float* __restrict__ scalar_agg,
    const unsigned short* __restrict__ W1T, const unsigned short* __restrict__ W2T,
    const float* __restrict__ b1, const float* __restrict__ b2,
    float* __restrict__ out, int Natoms) {
  __shared__ __align__(16) unsigned short sA[2][32][256];
  __shared__ __align__(16) unsigned short sH[2][32][128];
  const int lane = threadIdx.x & 31;
  const int w    = threadIdx.x >> 5;
  const int a0   = (blockIdx.x * 2 + w) * 32;
  if (a0 + 32 > Natoms) return;

  {
    const uint4* src = (const uint4*)(atom_bf + (size_t)a0 * H);
    uint4* dst = (uint4*)&sA[w][0][0];  // row stride = 256 halves = 32 uint4
    for (int t = lane; t < 512; t += 32) {
      int m = t >> 4, q = t & 15;
      dst[m * 32 + q] = src[m * 16 + q];
    }
  }
  for (int t = lane; t < 32 * H; t += 32) {
    int m = t >> 7, h = t & 127;
    size_t n = (size_t)(a0 + m);
    float f0 = force_node[n * 384 + h];
    float f1 = force_node[n * 384 + 128 + h];
    float f2 = force_node[n * 384 + 256 + h];
    sA[w][m][128 + h] = f2bf(scalar_agg[n * H + h] + f0 * f0 + f1 * f1 + f2 * f2);
  }
  lds_fence();

  const int lnN = lane & 15, hi8 = ((lane >> 4) & 1) * 8;
  v8f C[2][8];
  zeroC(C);
  gemm2tile<256>(&sA[w][0][0], 256, W1T, lane, C);
  silu_store(&sH[w][0][0], b1, C, lane);
  lds_fence();
  zeroC(C);
  gemm2tile<128>(&sH[w][0][0], 128, W2T, lane, C);
#pragma unroll
  for (int nt = 0; nt < 8; ++nt) {
    int ncol = nt * 16 + lnN;
    float bb = b2[ncol];
#pragma unroll
    for (int mt = 0; mt < 2; ++mt)
#pragma unroll
      for (int r = 0; r < 8; ++r) {
        size_t n = (size_t)(a0 + mt * 16 + r + hi8);
        out[n * H + ncol] = atom_node[n * H + ncol] + C[mt][nt][r] + bb;
      }
  }
}

// ---------------- force kernel ----------------
// rows fr = n*3 + d over (N,3); vec_agg is already laid out as (N*3, H).
__global__ void __launch_bounds__(64) force_kernel(
    const float* __restrict__ force_node, const unsigned short* __restrict__ atom_bf,
    const float* __restrict__ vec_agg,
    const unsigned short* __restrict__ W1T, const unsigned short* __restrict__ W2T,
    const float* __restrict__ b1, const float* __restrict__ b2,
    float* __restrict__ out_force, int Nrows) {
  __shared__ __align__(16) unsigned short sA[2][32][256];
  __shared__ __align__(16) unsigned short sH[2][32][128];
  const int lane = threadIdx.x & 31;
  const int w    = threadIdx.x >> 5;
  const int f0r  = (blockIdx.x * 2 + w) * 32;
  if (f0r + 32 > Nrows) return;

  for (int t = lane; t < 32 * H; t += 32) {
    int m = t >> 7, h = t & 127;
    int fr = f0r + m;
    int n  = fr / 3;
    sA[w][m][h]       = atom_bf[(size_t)n * H + h];
    sA[w][m][128 + h] = f2bf(vec_agg[(size_t)fr * H + h]);
  }
  lds_fence();

  const int lnN = lane & 15, hi8 = ((lane >> 4) & 1) * 8;
  v8f C[2][8];
  zeroC(C);
  gemm2tile<256>(&sA[w][0][0], 256, W1T, lane, C);
  silu_store(&sH[w][0][0], b1, C, lane);
  lds_fence();
  zeroC(C);
  gemm2tile<128>(&sH[w][0][0], 128, W2T, lane, C);
#pragma unroll
  for (int nt = 0; nt < 8; ++nt) {
    int ncol = nt * 16 + lnN;
    float bb = b2[ncol];
#pragma unroll
    for (int mt = 0; mt < 2; ++mt)
#pragma unroll
      for (int r = 0; r < 8; ++r) {
        size_t fr = (size_t)(f0r + mt * 16 + r + hi8);
        float f = force_node[fr * H + ncol];
        out_force[fr * H + ncol] = f * (1.0f + C[mt][nt][r] + bb);  // f + mlp*f
      }
  }
}

// ---------------- launcher ----------------
extern "C" void kernel_launch(void* const* d_in, const int* in_sizes, int n_in,
                              void* d_out, int out_size, void* d_ws, size_t ws_size,
                              hipStream_t stream) {
  const float* atom_node  = (const float*)d_in[0];
  const float* force_node = (const float*)d_in[1];
  const float* dir_edge   = (const float*)d_in[2];
  const float* dist_edge  = (const float*)d_in[3];
  const long long* edge_index = (const long long*)d_in[4];
  const float* W_msg1  = (const float*)d_in[5];  const float* b_msg1  = (const float*)d_in[6];
  const float* W_msg2  = (const float*)d_in[7];  const float* b_msg2  = (const float*)d_in[8];
  const float* W_vec1  = (const float*)d_in[9];  const float* b_vec1  = (const float*)d_in[10];
  const float* W_vec2  = (const float*)d_in[11]; const float* b_vec2  = (const float*)d_in[12];
  const float* W_node1 = (const float*)d_in[13]; const float* b_node1 = (const float*)d_in[14];
  const float* W_node2 = (const float*)d_in[15]; const float* b_node2 = (const float*)d_in[16];
  const float* W_force1 = (const float*)d_in[17]; const float* b_force1 = (const float*)d_in[18];
  const float* W_force2 = (const float*)d_in[19]; const float* b_force2 = (const float*)d_in[20];

  const int N = in_sizes[0] / H;     // 20000
  const int E = in_sizes[3] / RDIM;  // 640000

  // workspace layout
  float* scalar_agg = (float*)d_ws;                      // N*H f32
  float* vec_agg    = scalar_agg + (size_t)N * H;        // N*3*H f32
  unsigned short* atom_bf = (unsigned short*)(vec_agg + (size_t)N * 3 * H);  // N*H bf16
  unsigned short* wts  = atom_bf + (size_t)N * H;        // transposed bf16 weights
  unsigned short* WmT1 = wts;                  // 128 x 160
  unsigned short* WvT1 = WmT1 + 128 * 160;     // 128 x 160
  unsigned short* WmT2 = WvT1 + 128 * 160;     // 128 x 128
  unsigned short* WvT2 = WmT2 + 128 * 128;     // 128 x 128
  unsigned short* Wn1T = WvT2 + 128 * 128;     // 128 x 256
  unsigned short* Wn2T = Wn1T + 128 * 256;     // 128 x 128
  unsigned short* Wf1T = Wn2T + 128 * 128;     // 128 x 256
  unsigned short* Wf2T = Wf1T + 128 * 256;     // 128 x 128

  // 1) convert + transpose weights to bf16
  wt_convert<<<(160 * 128 + 255) / 256, 256, 0, stream>>>(W_msg1,  WmT1, 160, H);
  wt_convert<<<(160 * 128 + 255) / 256, 256, 0, stream>>>(W_vec1,  WvT1, 160, H);
  wt_convert<<<(128 * 128 + 255) / 256, 256, 0, stream>>>(W_msg2,  WmT2, 128, H);
  wt_convert<<<(128 * 128 + 255) / 256, 256, 0, stream>>>(W_vec2,  WvT2, 128, H);
  wt_convert<<<(256 * 128 + 255) / 256, 256, 0, stream>>>(W_node1, Wn1T, 256, H);
  wt_convert<<<(128 * 128 + 255) / 256, 256, 0, stream>>>(W_node2, Wn2T, 128, H);
  wt_convert<<<(256 * 128 + 255) / 256, 256, 0, stream>>>(W_force1, Wf1T, 256, H);
  wt_convert<<<(128 * 128 + 255) / 256, 256, 0, stream>>>(W_force2, Wf2T, 128, H);

  // 2) atom_node -> bf16 once (gathered per edge)
  a2bf<<<((size_t)N * H + 255) / 256, 256, 0, stream>>>(atom_node, atom_bf, N * H);

  // 3) zero L2-resident accumulators (scalar_agg + vec_agg contiguous)
  hipMemsetAsync(scalar_agg, 0, (size_t)N * H * 4 * sizeof(float), stream);

  // 4) edge message passing (WMMA GEMMs + atomic scatter)
  edge_kernel<<<(E + 63) / 64, 64, 0, stream>>>(
      force_node, dir_edge, dist_edge, edge_index, atom_bf,
      WmT1, WvT1, WmT2, WvT2, b_msg1, b_msg2, b_vec1, b_vec2,
      scalar_agg, vec_agg, E);

  // 5) node update
  float* out_atom  = (float*)d_out;
  float* out_force = out_atom + (size_t)N * H;
  node_kernel<<<(N + 63) / 64, 64, 0, stream>>>(
      atom_node, force_node, atom_bf, scalar_agg, Wn1T, Wn2T, b_node1, b_node2,
      out_atom, N);

  // 6) force update (3N rows)
  force_kernel<<<(3 * N + 63) / 64, 64, 0, stream>>>(
      force_node, atom_bf, vec_agg, Wf1T, Wf2T, b_force1, b_force2,
      out_force, 3 * N);
}
